// Contrastive_59983513256081
// MI455X (gfx1250) — compile-verified, matrix operand order checked by reading, not run on hardware
//
#include <hip/hip_runtime.h>
#include <cmath>

typedef _Float16 v16h __attribute__((ext_vector_type(16)));
typedef _Float16 v8h  __attribute__((ext_vector_type(8)));
typedef float    v8f  __attribute__((ext_vector_type(8)));

constexpr int NN = 8192;   // nodes
constexpr int DD = 512;    // hidden
constexpr int PP = 512;    // proj hidden
constexpr float TAU_INV = 2.0f;   // 1/0.5

// ---------------------------------------------------------------------------
// Fragment loader for 16-bit A (16xK tile slice) and B (Kx16 via row-major
// B-transposed source). Per CDNA5 ISA layout (CDNA5 ISA, 16-bit A 16x32):
//   lane m (0-15):  VGPR0-3 hold K = k0+{0..7},  VGPR4-7 hold K = k0+{16..23}
//   lane m+16:      VGPR0-3 hold K = k0+{8..15}, VGPR4-7 hold K = k0+{24..31}
// So each lane loads two contiguous 16-byte chunks from its row.
// ---------------------------------------------------------------------------
__device__ __forceinline__ v16h load_frag16(const _Float16* __restrict__ rowp,
                                            int k0, int hh) {
  const _Float16* p = rowp + k0 + 8 * hh;
  v8h lo = *reinterpret_cast<const v8h*>(p);
  v8h hi = *reinterpret_cast<const v8h*>(p + 16);
  v16h r;
#pragma unroll
  for (int i = 0; i < 8; ++i) { r[i] = lo[i]; r[i + 8] = hi[i]; }
  return r;
}

__device__ __forceinline__ v8f wmma_f16(v16h a, v16h b, v8f c) {
  return __builtin_amdgcn_wmma_f32_16x16x32_f16(
      /*neg_a=*/false, a, /*neg_b=*/false, b,
      /*c_mod=*/(short)0, c, /*reuse_a=*/false, /*reuse_b=*/false);
}

// ---------------------------------------------------------------------------
// Elementwise helpers
// ---------------------------------------------------------------------------
__global__ void k_f32_to_f16(const float* __restrict__ in,
                             _Float16* __restrict__ out, int n) {
  int i = blockIdx.x * blockDim.x + threadIdx.x;
  if (i < n) out[i] = (_Float16)in[i];
}

// in: [R][C] f32 row-major  ->  out: [C][R] f16 (transposed)
__global__ void k_transpose_to_f16(const float* __restrict__ in,
                                   _Float16* __restrict__ out, int R, int C) {
  int i = blockIdx.x * blockDim.x + threadIdx.x;
  if (i < R * C) {
    int r = i / C, c = i % C;
    out[(size_t)c * R + r] = (_Float16)in[i];
  }
}

__global__ void k_zero(float* __restrict__ p, int n) {
  int i = blockIdx.x * blockDim.x + threadIdx.x;
  if (i < n) p[i] = 0.0f;
}

// ---------------------------------------------------------------------------
// GEMM: O = act(A @ Bt^T + bias).  A:[M][K] f16, Bt:[Nout][K] f16.
// Block = 256 threads = 8 waves; each wave computes a 16x64 tile.
// grid = (M/128, Nout/64).
// ---------------------------------------------------------------------------
__global__ __launch_bounds__(256) void k_gemm_bias_elu(
    const _Float16* __restrict__ A, const _Float16* __restrict__ Bt,
    const float* __restrict__ bias, _Float16* __restrict__ O,
    int M, int Nout, int K) {
  const int lane = threadIdx.x & 31;
  const int wave = threadIdx.x >> 5;
  const int m = lane & 15;
  const int hh = lane >> 4;
  const int i0 = blockIdx.x * 128 + wave * 16;
  const int j0 = blockIdx.y * 64;

  const _Float16* ar  = A  + (size_t)(i0 + m) * K;
  const _Float16* br0 = Bt + (size_t)(j0 + m) * K;
  const _Float16* br1 = br0 + (size_t)16 * K;
  const _Float16* br2 = br0 + (size_t)32 * K;
  const _Float16* br3 = br0 + (size_t)48 * K;

  v8f c0 = {}, c1 = {}, c2 = {}, c3 = {};
  for (int k0 = 0; k0 < K; k0 += 32) {
    v16h a  = load_frag16(ar,  k0, hh);
    v16h b0 = load_frag16(br0, k0, hh);
    v16h b1 = load_frag16(br1, k0, hh);
    v16h b2 = load_frag16(br2, k0, hh);
    v16h b3 = load_frag16(br3, k0, hh);
    c0 = wmma_f16(a, b0, c0);
    c1 = wmma_f16(a, b1, c1);
    c2 = wmma_f16(a, b2, c2);
    c3 = wmma_f16(a, b3, c3);
  }
  v8f cc[4] = {c0, c1, c2, c3};
#pragma unroll
  for (int s = 0; s < 4; ++s) {
    const int col = j0 + s * 16 + m;
    const float bv = bias[col];
#pragma unroll
    for (int r = 0; r < 8; ++r) {
      const int row = i0 + r + 8 * hh;
      float x = cc[s][r] + bv;
      x = (x > 0.0f) ? x : expm1f(x);   // jax.nn.elu, alpha=1
      O[(size_t)row * Nout + col] = (_Float16)x;
    }
  }
}

__global__ __launch_bounds__(256) void k_gemm_bias_f32(
    const _Float16* __restrict__ A, const _Float16* __restrict__ Bt,
    const float* __restrict__ bias, float* __restrict__ O,
    int M, int Nout, int K) {
  const int lane = threadIdx.x & 31;
  const int wave = threadIdx.x >> 5;
  const int m = lane & 15;
  const int hh = lane >> 4;
  const int i0 = blockIdx.x * 128 + wave * 16;
  const int j0 = blockIdx.y * 64;

  const _Float16* ar  = A  + (size_t)(i0 + m) * K;
  const _Float16* br0 = Bt + (size_t)(j0 + m) * K;
  const _Float16* br1 = br0 + (size_t)16 * K;
  const _Float16* br2 = br0 + (size_t)32 * K;
  const _Float16* br3 = br0 + (size_t)48 * K;

  v8f c0 = {}, c1 = {}, c2 = {}, c3 = {};
  for (int k0 = 0; k0 < K; k0 += 32) {
    v16h a  = load_frag16(ar,  k0, hh);
    v16h b0 = load_frag16(br0, k0, hh);
    v16h b1 = load_frag16(br1, k0, hh);
    v16h b2 = load_frag16(br2, k0, hh);
    v16h b3 = load_frag16(br3, k0, hh);
    c0 = wmma_f16(a, b0, c0);
    c1 = wmma_f16(a, b1, c1);
    c2 = wmma_f16(a, b2, c2);
    c3 = wmma_f16(a, b3, c3);
  }
  v8f cc[4] = {c0, c1, c2, c3};
#pragma unroll
  for (int s = 0; s < 4; ++s) {
    const int col = j0 + s * 16 + m;
    const float bv = bias[col];
#pragma unroll
    for (int r = 0; r < 8; ++r) {
      const int row = i0 + r + 8 * hh;
      O[(size_t)row * Nout + col] = cc[s][r] + bv;
    }
  }
}

// ---------------------------------------------------------------------------
// Per-row: norms of h1,h2 -> write normalized f16 rows; exact f32 diag
// d12[i] = <h1_i, h2_i> / (||h1_i|| * ||h2_i||). One block (128 thr) per row.
// ---------------------------------------------------------------------------
__global__ __launch_bounds__(128) void k_normalize_diag(
    const float* __restrict__ h1, const float* __restrict__ h2,
    _Float16* __restrict__ h1n, _Float16* __restrict__ h2n,
    float* __restrict__ d12, int D) {
  __shared__ float sh0[128], sh1[128], sh2[128];
  const int tid = threadIdx.x;
  const size_t base = (size_t)blockIdx.x * D;
  float s1 = 0.f, s2 = 0.f, s12 = 0.f;
  for (int t = tid; t < D; t += 128) {
    float a = h1[base + t], b = h2[base + t];
    s1 += a * a; s2 += b * b; s12 += a * b;
  }
  sh0[tid] = s1; sh1[tid] = s2; sh2[tid] = s12;
  __syncthreads();
  for (int s = 64; s > 0; s >>= 1) {
    if (tid < s) { sh0[tid] += sh0[tid + s]; sh1[tid] += sh1[tid + s]; sh2[tid] += sh2[tid + s]; }
    __syncthreads();
  }
  const float n1 = fmaxf(sqrtf(sh0[0]), 1e-12f);
  const float n2 = fmaxf(sqrtf(sh1[0]), 1e-12f);
  const float i1 = 1.0f / n1, i2 = 1.0f / n2;
  for (int t = tid; t < D; t += 128) {
    h1n[base + t] = (_Float16)(h1[base + t] * i1);
    h2n[base + t] = (_Float16)(h2[base + t] * i2);
  }
  if (tid == 0) d12[blockIdx.x] = sh2[0] * i1 * i2;
}

// ---------------------------------------------------------------------------
// Fused Gram tile + exp + row/col-sum. S = A @ B^T (A,B: [n][k] f16, row-major
// normalized). Never materializes S: exponentiates tiles in registers and
// reduces to rowsum (always) and colsum (S12 only) via xor-shuffles + atomics.
// Each wave computes a 32x64 tile (2 A-frags + 4 B-frags -> 8 WMMAs per
// K-step: 0.75 fragment loads per WMMA). Block = 256 thr = 8 waves -> 256x64
// block tile. grid = (n/256, n/64).
// ---------------------------------------------------------------------------
__global__ __launch_bounds__(256) void k_gram(
    const _Float16* __restrict__ A, const _Float16* __restrict__ B,
    float* __restrict__ rowsum, float* __restrict__ colsum,
    int n, int k, int do_col) {
  const int lane = threadIdx.x & 31;
  const int wave = threadIdx.x >> 5;
  const int m = lane & 15;
  const int hh = lane >> 4;
  const int i0 = blockIdx.x * 256 + wave * 32;
  const int j0 = blockIdx.y * 64;

  const _Float16* ar0 = A + (size_t)(i0 + m) * k;
  const _Float16* ar1 = ar0 + (size_t)16 * k;
  const _Float16* br0 = B + (size_t)(j0 + m) * k;
  const _Float16* br1 = br0 + (size_t)16 * k;
  const _Float16* br2 = br0 + (size_t)32 * k;
  const _Float16* br3 = br0 + (size_t)48 * k;

  v8f c00 = {}, c01 = {}, c02 = {}, c03 = {};   // rows i0..i0+15
  v8f c10 = {}, c11 = {}, c12 = {}, c13 = {};   // rows i0+16..i0+31
  for (int k0 = 0; k0 < k; k0 += 32) {
    v16h a0 = load_frag16(ar0, k0, hh);
    v16h a1 = load_frag16(ar1, k0, hh);
    v16h b0 = load_frag16(br0, k0, hh);
    v16h b1 = load_frag16(br1, k0, hh);
    v16h b2 = load_frag16(br2, k0, hh);
    v16h b3 = load_frag16(br3, k0, hh);
    c00 = wmma_f16(a0, b0, c00);
    c01 = wmma_f16(a0, b1, c01);
    c02 = wmma_f16(a0, b2, c02);
    c03 = wmma_f16(a0, b3, c03);
    c10 = wmma_f16(a1, b0, c10);
    c11 = wmma_f16(a1, b1, c11);
    c12 = wmma_f16(a1, b2, c12);
    c13 = wmma_f16(a1, b3, c13);
  }

  // exp epilogue. C layout: (VGPR r, lane) -> row = rowbase + r + 8*hh,
  // col(subtile s) = j0 + 16*s + m.
  float rp0[8], rp1[8];
  float cp0 = 0.f, cp1 = 0.f, cp2 = 0.f, cp3 = 0.f;
#pragma unroll
  for (int r = 0; r < 8; ++r) {
    float e00 = __expf(c00[r] * TAU_INV);
    float e01 = __expf(c01[r] * TAU_INV);
    float e02 = __expf(c02[r] * TAU_INV);
    float e03 = __expf(c03[r] * TAU_INV);
    float e10 = __expf(c10[r] * TAU_INV);
    float e11 = __expf(c11[r] * TAU_INV);
    float e12 = __expf(c12[r] * TAU_INV);
    float e13 = __expf(c13[r] * TAU_INV);
    rp0[r] = e00 + e01 + e02 + e03;
    rp1[r] = e10 + e11 + e12 + e13;
    cp0 += e00 + e10;
    cp1 += e01 + e11;
    cp2 += e02 + e12;
    cp3 += e03 + e13;
  }
  // Row sums: xor 1,2,4,8 reduces within each 16-lane half (wave32).
#pragma unroll
  for (int r = 0; r < 8; ++r) {
    float v0 = rp0[r];
    v0 += __shfl_xor(v0, 1);
    v0 += __shfl_xor(v0, 2);
    v0 += __shfl_xor(v0, 4);
    v0 += __shfl_xor(v0, 8);
    float v1 = rp1[r];
    v1 += __shfl_xor(v1, 1);
    v1 += __shfl_xor(v1, 2);
    v1 += __shfl_xor(v1, 4);
    v1 += __shfl_xor(v1, 8);
    if (m == 0) {
      atomicAdd(rowsum + i0 + r + 8 * hh, v0);
      atomicAdd(rowsum + i0 + 16 + r + 8 * hh, v1);
    }
  }
  if (do_col) {
    // combine the two 8-row halves of each column, then one atomic per column
    cp0 += __shfl_xor(cp0, 16);
    cp1 += __shfl_xor(cp1, 16);
    cp2 += __shfl_xor(cp2, 16);
    cp3 += __shfl_xor(cp3, 16);
    if (hh == 0) {
      atomicAdd(colsum + j0 + m,      cp0);
      atomicAdd(colsum + j0 + 16 + m, cp1);
      atomicAdd(colsum + j0 + 32 + m, cp2);
      atomicAdd(colsum + j0 + 48 + m, cp3);
    }
  }
}

// ---------------------------------------------------------------------------
// loss_i = 0.5 * [ (log(rs11+rs12-e^2) - d12*2) + (log(rs22+cs12-e^2) - d12*2) ]
// out = mean_i(loss_i). diag(refl) = exp(1/tau) = e^2 (rows are unit-norm).
// ---------------------------------------------------------------------------
__global__ __launch_bounds__(256) void k_final(
    const float* __restrict__ d12, const float* __restrict__ rs11,
    const float* __restrict__ rs12, const float* __restrict__ cs12,
    const float* __restrict__ rs22, float* __restrict__ out, int n) {
  __shared__ float sh[256];
  const float ediag = __expf(2.0f);  // exp(1/tau)
  float acc = 0.f;
  for (int i = threadIdx.x; i < n; i += 256) {
    float dt = d12[i] * TAU_INV;
    float l1 = __logf(fmaxf(rs11[i] + rs12[i] - ediag, 1e-30f)) - dt;
    float l2 = __logf(fmaxf(rs22[i] + cs12[i] - ediag, 1e-30f)) - dt;
    acc += 0.5f * (l1 + l2);
  }
  sh[threadIdx.x] = acc;
  __syncthreads();
  for (int s = 128; s > 0; s >>= 1) {
    if (threadIdx.x < s) sh[threadIdx.x] += sh[threadIdx.x + s];
    __syncthreads();
  }
  if (threadIdx.x == 0) out[0] = sh[0] / (float)n;
}

// ---------------------------------------------------------------------------
extern "C" void kernel_launch(void* const* d_in, const int* in_sizes, int n_in,
                              void* d_out, int out_size, void* d_ws, size_t ws_size,
                              hipStream_t stream) {
  const float* z1    = (const float*)d_in[0];
  const float* z2    = (const float*)d_in[1];
  // d_in[2] = index (arange; unique() identity path -> unused)
  const float* fc1_w = (const float*)d_in[3];  // [D][P]
  const float* fc1_b = (const float*)d_in[4];  // [P]
  const float* fc2_w = (const float*)d_in[5];  // [P][D]
  const float* fc2_b = (const float*)d_in[6];  // [D]
  float* out = (float*)d_out;

  char* w = (char*)d_ws;
  size_t off = 0;
  auto take = [&](size_t bytes) -> char* {
    char* p = w + off;
    off += (bytes + 255) & ~(size_t)255;
    return p;
  };

  _Float16* z1h = (_Float16*)take((size_t)NN * DD * 2);
  _Float16* z2h = (_Float16*)take((size_t)NN * DD * 2);
  _Float16* w1t = (_Float16*)take((size_t)DD * PP * 2);  // fc1_w^T: [P][D]
  _Float16* w2t = (_Float16*)take((size_t)PP * DD * 2);  // fc2_w^T: [D][P]
  _Float16* e1  = (_Float16*)take((size_t)NN * PP * 2);
  _Float16* e2  = (_Float16*)take((size_t)NN * PP * 2);
  float*    h1  = (float*)take((size_t)NN * DD * 4);
  float*    h2  = (float*)take((size_t)NN * DD * 4);
  float*    sums = (float*)take((size_t)4 * NN * 4);     // rs11|rs12|cs12|rs22
  float*    d12  = (float*)take((size_t)NN * 4);
  // z1h/z2h are dead after the fc1 GEMMs -> reuse as normalized f16 buffers
  _Float16* h1n = z1h;
  _Float16* h2n = z2h;
  float* rs11 = sums;
  float* rs12 = sums + NN;
  float* cs12 = sums + 2 * NN;
  float* rs22 = sums + 3 * NN;

  // 1) precision conversion + weight transposes + accumulator zeroing
  const int ne = NN * DD;
  k_f32_to_f16<<<(ne + 255) / 256, 256, 0, stream>>>(z1, z1h, ne);
  k_f32_to_f16<<<(ne + 255) / 256, 256, 0, stream>>>(z2, z2h, ne);
  k_transpose_to_f16<<<(DD * PP + 255) / 256, 256, 0, stream>>>(fc1_w, w1t, DD, PP);
  k_transpose_to_f16<<<(PP * DD + 255) / 256, 256, 0, stream>>>(fc2_w, w2t, PP, DD);
  k_zero<<<(4 * NN + 255) / 256, 256, 0, stream>>>(sums, 4 * NN);

  // 2) projection MLP (WMMA f16): e = elu(z @ fc1 + b1); h = e @ fc2 + b2
  dim3 g1(NN / 128, PP / 64);
  k_gemm_bias_elu<<<g1, 256, 0, stream>>>(z1h, w1t, fc1_b, e1, NN, PP, DD);
  k_gemm_bias_elu<<<g1, 256, 0, stream>>>(z2h, w1t, fc1_b, e2, NN, PP, DD);
  dim3 g2(NN / 128, DD / 64);
  k_gemm_bias_f32<<<g2, 256, 0, stream>>>(e1, w2t, fc2_b, h1, NN, DD, PP);
  k_gemm_bias_f32<<<g2, 256, 0, stream>>>(e2, w2t, fc2_b, h2, NN, DD, PP);

  // 3) row-normalize (f16 out) + exact f32 S12 diagonal
  k_normalize_diag<<<NN, 128, 0, stream>>>(h1, h2, h1n, h2n, d12, DD);

  // 4) fused Gram + exp + reductions (never materialize 8192x8192)
  dim3 gg(NN / 256, NN / 64);
  k_gram<<<gg, 256, 0, stream>>>(h1n, h1n, rs11, nullptr, NN, DD, 0);
  k_gram<<<gg, 256, 0, stream>>>(h1n, h2n, rs12, cs12,   NN, DD, 1);
  k_gram<<<gg, 256, 0, stream>>>(h2n, h2n, rs22, nullptr, NN, DD, 0);

  // 5) final log/mean reduce -> scalar
  k_final<<<1, 256, 0, stream>>>(d12, rs11, rs12, cs12, rs22, out, NN);
}
